// DEC_2619930050954
// MI455X (gfx1250) — matrix operand order repaired and checked by reading, not compiled, not run on hardware
//
#include <hip/hip_runtime.h>
#include <hip/hip_bf16.h>

typedef __attribute__((ext_vector_type(16))) _Float16 v16h;
typedef __attribute__((ext_vector_type(8)))  _Float16 v8h;
typedef __attribute__((ext_vector_type(8)))  float    v8f;

union Frag16 { v16h v; v8h h[2]; };

#define BATCH 512
#define SEQL  100
#define HID   256
#define G3    768
#define WINN  5
#define DD    10

// ---------------------------------------------------------------------------
// f32 -> f16 weight conversion
// ---------------------------------------------------------------------------
__global__ __launch_bounds__(256) void cvt_f32_to_f16(const float* __restrict__ src,
                                                      _Float16* __restrict__ dst, int n) {
    int i = blockIdx.x * 256 + threadIdx.x;
    if (i < n) dst[i] = (_Float16)src[i];
}

// ---------------------------------------------------------------------------
// xw0[m, g] = bih0[g] + sum_{i<3} x[m*3+i] * Wih0[g*3+i]   (K=3, trivial)
// ---------------------------------------------------------------------------
__global__ __launch_bounds__(256) void xw0_kernel(const float* __restrict__ x,
                                                  const float* __restrict__ Wih0,
                                                  const float* __restrict__ bih0,
                                                  float* __restrict__ xw) {
    long long idx = (long long)blockIdx.x * 256 + threadIdx.x;
    if (idx >= (long long)BATCH * SEQL * G3) return;
    int g = (int)(idx % G3);
    long long m = idx / G3;
    float a = bih0[g];
    a += x[m * 3 + 0] * Wih0[g * 3 + 0];
    a += x[m * 3 + 1] * Wih0[g * 3 + 1];
    a += x[m * 3 + 2] * Wih0[g * 3 + 2];
    xw[idx] = a;
}

// ---------------------------------------------------------------------------
// Batch-parallel GRU scan. One workgroup owns 16 batch rows; h state lives in
// LDS (16x256 f16). 8 waves x 2 hidden tiles x 3 gates; K=256 via 8 WMMA ops
// of v_wmma_f32_16x16x32_f16 per tile per step. No inter-block sync needed.
// ---------------------------------------------------------------------------
__global__ __launch_bounds__(256) void gru_scan(const float* __restrict__ xw,      // (B,L,3H)
                                                const _Float16* __restrict__ whh,  // (3H,H) f16
                                                const float* __restrict__ bhh,     // (3H)
                                                float* __restrict__ out_f32,       // (B,L,H)
                                                _Float16* __restrict__ out_f16) {  // (B,L,H)
    __shared__ _Float16 hbuf[16 * HID];  // 8KB

    const int wave  = threadIdx.x >> 5;
    const int lane  = threadIdx.x & 31;
    const int b0    = blockIdx.x * 16;
    const int lhalf = lane >> 4;        // 0: lanes 0-15, 1: lanes 16-31
    const int mrow  = lane & 15;        // A-fragment row
    const int ncol  = lane & 15;        // C/D column within tile
    const int crow0 = lhalf * 8;        // C/D row base
    const int n_t0  = wave * 32;        // hidden-col base (2 tiles of 16 per wave)

    for (int i = threadIdx.x; i < 16 * HID; i += 256) hbuf[i] = (_Float16)0.0f;
    __syncthreads();

    // hoist per-lane bias values (gate g, tile t)
    float bh[2][3];
#pragma unroll
    for (int t = 0; t < 2; ++t)
#pragma unroll
        for (int g = 0; g < 3; ++g)
            bh[t][g] = bhh[g * HID + n_t0 + t * 16 + ncol];

    for (int l = 0; l < SEQL; ++l) {
        v8f zero = {};
        v8f acc[2][3];
#pragma unroll
        for (int t = 0; t < 2; ++t)
#pragma unroll
            for (int g = 0; g < 3; ++g) acc[t][g] = zero;

#pragma unroll
        for (int k = 0; k < 8; ++k) {  // K = 8 * 32 = 256
            Frag16 a;
            const _Float16* ap = &hbuf[mrow * HID + k * 32 + lhalf * 8];
            a.h[0] = *(const v8h*)ap;
            a.h[1] = *(const v8h*)(ap + 16);
#pragma unroll
            for (int t = 0; t < 2; ++t) {
#pragma unroll
                for (int g = 0; g < 3; ++g) {
                    Frag16 bf;
                    const _Float16* bp =
                        whh + (size_t)(g * HID + n_t0 + t * 16 + ncol) * HID + k * 32 + lhalf * 8;
                    bf.h[0] = *(const v8h*)bp;
                    bf.h[1] = *(const v8h*)(bp + 16);
                    acc[t][g] = __builtin_amdgcn_wmma_f32_16x16x32_f16(
                        false, a.v, false, bf.v, (short)0, acc[t][g], false, false);
                }
            }
        }
        __syncthreads();  // all waves done reading hbuf (A-fragments)

#pragma unroll
        for (int t = 0; t < 2; ++t) {
            const int j = n_t0 + t * 16 + ncol;  // hidden index (disjoint across waves)
#pragma unroll
            for (int r = 0; r < 8; ++r) {
                const int m = crow0 + r;
                const size_t xbase = ((size_t)(b0 + m) * SEQL + l) * G3 + j;
                const float xr = xw[xbase];
                const float xz = xw[xbase + HID];
                const float xn = xw[xbase + 2 * HID];
                const float hr = acc[t][0][r] + bh[t][0];
                const float hz = acc[t][1][r] + bh[t][1];
                const float hn = acc[t][2][r] + bh[t][2];
                const float hold = (float)hbuf[m * HID + j];
                const float rg = 1.0f / (1.0f + __expf(-(xr + hr)));
                const float zg = 1.0f / (1.0f + __expf(-(xz + hz)));
                const float ng = tanhf(xn + rg * hn);
                const float hnew = (1.0f - zg) * ng + zg * hold;
                hbuf[m * HID + j] = (_Float16)hnew;  // own cell only: no cross-lane hazard
                const size_t obase = ((size_t)(b0 + m) * SEQL + l) * HID + j;
                out_f32[obase] = hnew;
                out_f16[obase] = (_Float16)hnew;
            }
        }
        __syncthreads();  // h fully updated before next step's A-fragment loads
    }
}

// ---------------------------------------------------------------------------
// Generic WMMA GEMM with bias: C[orow, n] = bias[n] + sum_k A[m,k] * W[n,k]
// A: (M,K) f16 row-major, W: (N,K) f16 row-major. One 16x16 tile per wave.
// remap95: orow = (m/95)*100 + (m%95) + 5 (branch FC writes into out[:,5:,:]).
// ---------------------------------------------------------------------------
__global__ __launch_bounds__(256) void wmma_gemm_bias(const _Float16* __restrict__ A,
                                                      const _Float16* __restrict__ W,
                                                      const float* __restrict__ bias,
                                                      float* __restrict__ C,
                                                      int K, int N, int ldc, int remap95) {
    const int wave  = threadIdx.x >> 5;
    const int lane  = threadIdx.x & 31;
    const int mt    = blockIdx.x;
    const int nt    = blockIdx.y * 8 + wave;
    if (nt * 16 >= N) return;  // wave-uniform
    const int lhalf = lane >> 4;
    const int mr    = lane & 15;
    const int nc    = lane & 15;

    v8f acc = {};
    for (int k = 0; k < K; k += 32) {
        Frag16 a, bf;
        const _Float16* ap = A + (size_t)(mt * 16 + mr) * K + k + lhalf * 8;
        a.h[0] = *(const v8h*)ap;
        a.h[1] = *(const v8h*)(ap + 16);
        const _Float16* bp = W + (size_t)(nt * 16 + nc) * K + k + lhalf * 8;
        bf.h[0] = *(const v8h*)bp;
        bf.h[1] = *(const v8h*)(bp + 16);
        acc = __builtin_amdgcn_wmma_f32_16x16x32_f16(false, a.v, false, bf.v, (short)0, acc,
                                                     false, false);
    }
    const int n  = nt * 16 + nc;
    const float bn = bias[n];
#pragma unroll
    for (int r = 0; r < 8; ++r) {
        const int m = mt * 16 + lhalf * 8 + r;
        const int orow = remap95 ? ((m / 95) * SEQL + (m % 95) + WINN) : m;
        C[(size_t)orow * ldc + n] = acc[r] + bn;
    }
}

// ---------------------------------------------------------------------------
// q/e attention projections: q[m] = out[m,:] . attn_w[:H], e[m] = . attn_w[H:]
// ---------------------------------------------------------------------------
__global__ __launch_bounds__(256) void qe_kernel(const float* __restrict__ out,
                                                 const float* __restrict__ attn_w,
                                                 float* __restrict__ q, float* __restrict__ e) {
    int m = blockIdx.x * 256 + threadIdx.x;
    if (m >= BATCH * SEQL) return;
    float aq = 0.f, ae = 0.f;
    const float* row = out + (size_t)m * HID;
    for (int h = 0; h < HID; ++h) {
        aq += row[h] * attn_w[h];
        ae += row[h] * attn_w[HID + h];
    }
    q[m] = aq;
    e[m] = ae;
}

// ---------------------------------------------------------------------------
// softmax over WIN=5 + context; writes fused = concat(c, out[:,WIN:]) in f16.
// One block per (b,l); thread = hidden channel.
// ---------------------------------------------------------------------------
__global__ __launch_bounds__(256) void fused_kernel(const float* __restrict__ out,
                                                    const float* __restrict__ q,
                                                    const float* __restrict__ e,
                                                    _Float16* __restrict__ fused) {
    const int bl = blockIdx.x;           // 0 .. B*95-1
    const int b  = bl / (SEQL - WINN);
    const int l  = bl % (SEQL - WINN);
    const int h  = threadIdx.x;

    const float qv = q[b * SEQL + l + WINN];
    float s[WINN];
    float mx = -1e30f;
#pragma unroll
    for (int w = 0; w < WINN; ++w) {
        s[w] = qv + e[b * SEQL + l + w];
        mx = fmaxf(mx, s[w]);
    }
    float den = 0.f;
#pragma unroll
    for (int w = 0; w < WINN; ++w) {
        s[w] = __expf(s[w] - mx);
        den += s[w];
    }
    const float inv = 1.0f / den;
    float c = 0.f;
#pragma unroll
    for (int w = 0; w < WINN; ++w)
        c += s[w] * inv * out[((size_t)b * SEQL + l + w) * HID + h];

    fused[(size_t)bl * (2 * HID) + h]       = (_Float16)c;
    fused[(size_t)bl * (2 * HID) + HID + h] =
        (_Float16)out[((size_t)b * SEQL + l + WINN) * HID + h];
}

// ---------------------------------------------------------------------------
// decode: sigmoid(rnn1[b,l] . outW[:H] + rnn2[b,min(l+D,L-1)] . outW[H:] + b)
// ---------------------------------------------------------------------------
__global__ __launch_bounds__(256) void decode_kernel(const float* __restrict__ rnn1,
                                                     const float* __restrict__ rnn2,
                                                     const float* __restrict__ out_W,
                                                     const float* __restrict__ out_b,
                                                     float* __restrict__ dec) {
    int idx = blockIdx.x * 256 + threadIdx.x;
    if (idx >= BATCH * SEQL) return;
    const int b = idx / SEQL;
    const int l = idx % SEQL;
    const int l2 = (l + DD < SEQL) ? (l + DD) : (SEQL - 1);
    const float* r1 = rnn1 + ((size_t)b * SEQL + l) * HID;
    const float* r2 = rnn2 + ((size_t)b * SEQL + l2) * HID;
    float a = out_b[0];
    for (int h = 0; h < HID; ++h) {
        a += r1[h] * out_W[h];
        a += r2[h] * out_W[HID + h];
    }
    dec[idx] = 1.0f / (1.0f + __expf(-a));
}

// ---------------------------------------------------------------------------
extern "C" void kernel_launch(void* const* d_in, const int* in_sizes, int n_in,
                              void* d_out, int out_size, void* d_ws, size_t ws_size,
                              hipStream_t stream) {
    (void)in_sizes; (void)n_in; (void)out_size; (void)ws_size;
    const float* received  = (const float*)d_in[0];
    const float* Wih0[2] = {(const float*)d_in[1], (const float*)d_in[9]};
    const float* Whh0[2] = {(const float*)d_in[2], (const float*)d_in[10]};
    const float* bih0[2] = {(const float*)d_in[3], (const float*)d_in[11]};
    const float* bhh0[2] = {(const float*)d_in[4], (const float*)d_in[12]};
    const float* Wih1[2] = {(const float*)d_in[5], (const float*)d_in[13]};
    const float* Whh1[2] = {(const float*)d_in[6], (const float*)d_in[14]};
    const float* bih1[2] = {(const float*)d_in[7], (const float*)d_in[15]};
    const float* bhh1[2] = {(const float*)d_in[8], (const float*)d_in[16]};
    const float* attn_w = (const float*)d_in[17];
    const float* fc_W   = (const float*)d_in[18];
    const float* fc_b   = (const float*)d_in[19];
    const float* out_W  = (const float*)d_in[20];
    const float* out_b  = (const float*)d_in[21];

    char* ws = (char*)d_ws;
    size_t off = 0;
    auto take = [&](size_t bytes) -> char* {
        char* p = ws + off;
        off += (bytes + 255) & ~(size_t)255;
        return p;
    };
    float*     xw      = (float*)take((size_t)BATCH * SEQL * G3 * 4);        // 157 MB (reused)
    float*     outr[2] = {(float*)take((size_t)BATCH * SEQL * HID * 4),
                          (float*)take((size_t)BATCH * SEQL * HID * 4)};     // 52 MB x2
    _Float16*  l0h     = (_Float16*)take((size_t)BATCH * SEQL * HID * 2);    // 26 MB (reused)
    _Float16*  fusedb  = (_Float16*)take((size_t)BATCH * (SEQL - WINN) * 2 * HID * 2);
    float*     qbuf    = (float*)take((size_t)BATCH * SEQL * 4);
    float*     ebuf    = (float*)take((size_t)BATCH * SEQL * 4);
    _Float16*  whh0h[2] = {(_Float16*)take(G3 * HID * 2), (_Float16*)take(G3 * HID * 2)};
    _Float16*  whh1h[2] = {(_Float16*)take(G3 * HID * 2), (_Float16*)take(G3 * HID * 2)};
    _Float16*  wih1h[2] = {(_Float16*)take(G3 * HID * 2), (_Float16*)take(G3 * HID * 2)};
    _Float16*  fcwh     = (_Float16*)take(HID * 2 * HID * 2);

    const int NW = G3 * HID;  // 196608
    for (int r = 0; r < 2; ++r) {
        cvt_f32_to_f16<<<(NW + 255) / 256, 256, 0, stream>>>(Whh0[r], whh0h[r], NW);
        cvt_f32_to_f16<<<(NW + 255) / 256, 256, 0, stream>>>(Whh1[r], whh1h[r], NW);
        cvt_f32_to_f16<<<(NW + 255) / 256, 256, 0, stream>>>(Wih1[r], wih1h[r], NW);
    }
    cvt_f32_to_f16<<<(HID * 2 * HID + 255) / 256, 256, 0, stream>>>(fc_W, fcwh, HID * 2 * HID);

    for (int r = 0; r < 2; ++r) {
        // layer 0: input projection (K=3) then scan
        {
            long long n = (long long)BATCH * SEQL * G3;
            xw0_kernel<<<(int)((n + 255) / 256), 256, 0, stream>>>(received, Wih0[r], bih0[r], xw);
        }
        gru_scan<<<BATCH / 16, 256, 0, stream>>>(xw, whh0h[r], bhh0[r], outr[r], l0h);
        // layer 1 input GEMM: (51200 x 768) = l0h (51200 x 256) @ Wih1^T, +bih1
        {
            dim3 g(BATCH * SEQL / 16, G3 / (16 * 8));  // (3200, 6)
            wmma_gemm_bias<<<g, 256, 0, stream>>>(l0h, wih1h[r], bih1[r], xw, HID, G3, G3, 0);
        }
        gru_scan<<<BATCH / 16, 256, 0, stream>>>(xw, whh1h[r], bhh1[r], outr[r], l0h);
    }

    for (int r = 0; r < 2; ++r) {
        qe_kernel<<<(BATCH * SEQL + 255) / 256, 256, 0, stream>>>(outr[r], attn_w, qbuf, ebuf);
        fused_kernel<<<BATCH * (SEQL - WINN), 256, 0, stream>>>(outr[r], qbuf, ebuf, fusedb);
        // FC GEMM: (48640 x 256) = fused (48640 x 512) @ fc_W^T, +fc_b,
        // stored in-place into outr[r][:, WIN:, :] via the 95->100 row remap.
        dim3 g(BATCH * (SEQL - WINN) / 16, HID / (16 * 8));  // (3040, 2)
        wmma_gemm_bias<<<g, 256, 0, stream>>>(fusedb, fcwh, fc_b, outr[r], 2 * HID, HID, HID, 1);
    }

    decode_kernel<<<(BATCH * SEQL + 255) / 256, 256, 0, stream>>>(outr[0], outr[1], out_W, out_b,
                                                                  (float*)d_out);
}
